// MultiHeadAttention_91199335563986
// MI455X (gfx1250) — compile-verified, matrix operand order checked by reading, not run on hardware
//
#include <hip/hip_runtime.h>
#include <hip/hip_bf16.h>

// ---------------------------------------------------------------- constants
#define D_MODEL 1024
#define HEADS   16
#define D_HEAD  64
#define BB      2
#define SS      2048
#define M_TOT   (BB * SS)       // 4096
#define NEG_BIG -1e9f
#define LN_EPS  1e-6f

// GEMM tiling
#define BM 128
#define BN 128
#define BK 32
#define LDT 40                  // LDS row stride in halves (padded, conflict-free)
#define LDK 72                  // K-tile LDS stride in halves (conflict-free)

typedef __attribute__((ext_vector_type(16))) _Float16 v16h;
typedef __attribute__((ext_vector_type(4)))  _Float16 v4h;
typedef __attribute__((ext_vector_type(8)))  float    v8f;
typedef __attribute__((ext_vector_type(4)))  unsigned int v4u;
typedef __attribute__((ext_vector_type(4)))  int      v4i;

// CDNA5 async global->LDS staging (guarded; falls back to VGPR copy)
#if defined(__gfx1250__) && __has_builtin(__builtin_amdgcn_global_load_async_to_lds_b128) && \
    __has_builtin(__builtin_amdgcn_s_wait_asynccnt)
#define USE_ASYNC_LDS 1
#define AS1P __attribute__((address_space(1))) v4i*
#define AS3P __attribute__((address_space(3))) v4i*
#else
#define USE_ASYNC_LDS 0
#endif

union Frag16 { v16h h; v4u u[2]; };

// ---------------------------------------------------------------- helpers
__device__ __forceinline__ v8f wmma_f16(v16h a, v16h b, v8f c) {
  // D = A(16x32 f16) * B(32x16 f16) + C(16x16 f32)
  return __builtin_amdgcn_wmma_f32_16x16x32_f16(false, a, false, b, (short)0, c,
                                                false, false);
}

// Load a 16x32 f16 A/B fragment from a row-major half array.
// ISA layout: lane L (r=L&15, hi=L>>4) holds row (row0+r);
// halves 0..7 = cols [kb + hi*8 ..], halves 8..15 = cols [kb+16+hi*8 ..].
__device__ __forceinline__ v16h frag_ld(const _Float16* p, int stride, int row0, int kb) {
  const int lane = threadIdx.x & 31;
  const int r = lane & 15, hi = lane >> 4;
  const _Float16* rp = p + (size_t)(row0 + r) * stride + kb + hi * 8;
  Frag16 f;
  f.u[0] = *(const v4u*)(rp);
  f.u[1] = *(const v4u*)(rp + 16);
  return f.h;
}

__device__ __forceinline__ float rmax16(float x) {
#pragma unroll
  for (int m = 1; m < 16; m <<= 1) x = fmaxf(x, __shfl_xor(x, m, 32));
  return x;
}
__device__ __forceinline__ float rsum16(float x) {
#pragma unroll
  for (int m = 1; m < 16; m <<= 1) x += __shfl_xor(x, m, 32);
  return x;
}

// ---------------------------------------------------------------- kernel 1
// Y = X @ W^T with f16 WMMA, f32 accum. blockIdx.z selects Q/K/V.
// mode 0/1: write f16 (B,H,S,d).  mode 2: write f16 (B,H,d,S) (V transposed).
__global__ __launch_bounds__(256) void qkv_gemm_kernel(
    const float* __restrict__ xq, const float* __restrict__ xk,
    const float* __restrict__ xv, const float* __restrict__ Wq,
    const float* __restrict__ Wk, const float* __restrict__ Wv,
    _Float16* __restrict__ qh, _Float16* __restrict__ kh,
    _Float16* __restrict__ vhT) {
  __shared__ _Float16 As[BM * LDT];
  __shared__ _Float16 Bs[BN * LDT];

  const int mode = blockIdx.z;
  const float* X = (mode == 0) ? xq : (mode == 1) ? xk : xv;
  const float* W = (mode == 0) ? Wq : (mode == 1) ? Wk : Wv;

  const int tid = threadIdx.x;
  const int wave = tid >> 5, lane = tid & 31;
  const int wm = wave >> 1, wn = wave & 1;
  const int r = lane & 15, hi = lane >> 4;
  const int m0 = blockIdx.x * BM, n0 = blockIdx.y * BN;

  v8f acc[2][4] = {};
  v4h ra[4], rb[4];

  // register-staged global loads (pipeline: load k+1 while computing k)
  auto loadAB = [&](int k0) {
#pragma unroll
    for (int t = 0; t < 4; ++t) {
      const int c = tid + t * 256;         // 0..1023 float4 chunks
      const int row = c >> 3;
      const int c4 = (c & 7) * 4;
      const float4 a4 = *(const float4*)(X + (size_t)(m0 + row) * D_MODEL + k0 + c4);
      v4h ah = { (_Float16)a4.x, (_Float16)a4.y, (_Float16)a4.z, (_Float16)a4.w };
      ra[t] = ah;
      const float4 b4 = *(const float4*)(W + (size_t)(n0 + row) * D_MODEL + k0 + c4);
      v4h bh = { (_Float16)b4.x, (_Float16)b4.y, (_Float16)b4.z, (_Float16)b4.w };
      rb[t] = bh;
    }
  };

  loadAB(0);
  for (int k0 = 0; k0 < D_MODEL; k0 += BK) {
    __syncthreads();
#pragma unroll
    for (int t = 0; t < 4; ++t) {
      const int c = tid + t * 256;
      const int row = c >> 3;
      const int c4 = (c & 7) * 4;
      *(v4h*)(As + row * LDT + c4) = ra[t];
      *(v4h*)(Bs + row * LDT + c4) = rb[t];
    }
    __syncthreads();
    if (k0 + BK < D_MODEL) loadAB(k0 + BK);

    v16h a0 = frag_ld(As, LDT, wm * 32, 0);
    v16h a1 = frag_ld(As, LDT, wm * 32 + 16, 0);
#pragma unroll
    for (int j = 0; j < 4; ++j) {
      v16h bf = frag_ld(Bs, LDT, wn * 64 + j * 16, 0);
      acc[0][j] = wmma_f16(a0, bf, acc[0][j]);
      acc[1][j] = wmma_f16(a1, bf, acc[1][j]);
    }
  }

  // branch-free epilogue: per-mode dst + strides are block-uniform (SGPRs)
  _Float16* dst = (mode == 0) ? qh : (mode == 1) ? kh : vhT;
  const int sstr = (mode == 2) ? 1 : D_HEAD;     // stride of token index s
  const int dstr = (mode == 2) ? SS : 1;         // stride of head-dim index dh
#pragma unroll
  for (int i = 0; i < 2; ++i)
#pragma unroll
    for (int j = 0; j < 4; ++j)
#pragma unroll
      for (int e = 0; e < 8; ++e) {
        const int m = m0 + wm * 32 + i * 16 + e + 8 * hi;   // token row
        const int n = n0 + wn * 64 + j * 16 + r;            // feature col
        const int b = m >> 11, s = m & (SS - 1);
        const int h = n >> 6, dh = n & (D_HEAD - 1);
        dst[(size_t)b * (HEADS * SS * D_HEAD) + (size_t)h * (SS * D_HEAD) +
            (size_t)s * sstr + (size_t)dh * dstr] = (_Float16)acc[i][j][e];
      }
}

// ---------------------------------------------------------------- kernel 2
// Flash attention. Block = 8 waves = 128 q rows of one (b,h).
// K/V tiles staged cooperatively in LDS once per block; staging uses the
// CDNA5 async global->LDS path when the toolchain exposes it.
__global__ __launch_bounds__(256) void attn_kernel(
    const _Float16* __restrict__ qh, const _Float16* __restrict__ kh,
    const _Float16* __restrict__ vhT, const int* __restrict__ sen_len,
    _Float16* __restrict__ ctx) {
  __shared__ _Float16 Ks[32 * LDK];     // 32 keys x 64 d
  __shared__ _Float16 Vs[64 * LDT];     // 64 d x 32 keys (V^T tile)
  __shared__ _Float16 Pt[8][16 * 32];   // per-wave P tile

  const int bhead = blockIdx.y;                 // b*HEADS + h
  const int b = bhead >> 4, h = bhead & 15;
  const int tid = threadIdx.x;
  const int wave = tid >> 5, lane = tid & 31;
  const int r = lane & 15, hi = lane >> 4;
  const int q0b = blockIdx.x * 128;
  const int q0 = q0b + wave * 16;
  const int sl = sen_len[b];
  const int qmaxb = q0b + 128;
  const int nk = (qmaxb < sl) ? qmaxb : sl;     // block-uniform key bound

  const _Float16* qbase = qh + (size_t)bhead * SS * D_HEAD;
  const _Float16* kbase = kh + (size_t)bhead * SS * D_HEAD;
  const _Float16* vbase = vhT + (size_t)bhead * D_HEAD * SS;
  _Float16* pw = &Pt[wave][0];

  const v16h aq0 = frag_ld(qbase, D_HEAD, q0, 0);
  const v16h aq1 = frag_ld(qbase, D_HEAD, q0, 32);

  v8f cacc[4] = {};
  float rmax[8], rsum[8];
#pragma unroll
  for (int e = 0; e < 8; ++e) { rmax[e] = -1e30f; rsum[e] = 0.f; }

  // staging coords (one 16B K chunk + one 16B V chunk per thread per tile)
  const int krow = tid >> 3, kc = (tid & 7) * 8;    // K: 32 rows x 8 chunks
  const int vrow = tid >> 2, vc = (tid & 3) * 8;    // V: 64 rows x 4 chunks

  for (int kt = 0; kt < nk; kt += 32) {
    __syncthreads();   // protect previous tile's fragment reads
#if USE_ASYNC_LDS
    __builtin_amdgcn_global_load_async_to_lds_b128(
        (AS1P)(void*)(kbase + (size_t)(kt + krow) * D_HEAD + kc),
        (AS3P)(void*)(Ks + krow * LDK + kc), 0, 0);
    __builtin_amdgcn_global_load_async_to_lds_b128(
        (AS1P)(void*)(vbase + (size_t)vrow * SS + kt + vc),
        (AS3P)(void*)(Vs + vrow * LDT + vc), 0, 0);
    __builtin_amdgcn_s_wait_asynccnt(0);
#else
    *(v4u*)(Ks + krow * LDK + kc) =
        *(const v4u*)(kbase + (size_t)(kt + krow) * D_HEAD + kc);
    *(v4u*)(Vs + vrow * LDT + vc) =
        *(const v4u*)(vbase + (size_t)vrow * SS + kt + vc);
#endif
    __syncthreads();

    const v8f z = {};
    v8f s0 = wmma_f16(aq0, frag_ld(Ks, LDK, 0, 0), z);
    s0     = wmma_f16(aq1, frag_ld(Ks, LDK, 0, 32), s0);
    v8f s1 = wmma_f16(aq0, frag_ld(Ks, LDK, 16, 0), z);
    s1     = wmma_f16(aq1, frag_ld(Ks, LDK, 16, 32), s1);

#pragma unroll
    for (int e = 0; e < 8; ++e) {
      const int qrow = q0 + e + 8 * hi;
      const int j0 = kt + r, j1 = kt + 16 + r;
      float v0 = s0[e] * 0.125f;                      // 1/sqrt(64)
      float v1 = s1[e] * 0.125f;
      if (!((j0 <= qrow) && (j0 < sl))) v0 = NEG_BIG;
      if (!((j1 <= qrow) && (j1 < sl))) v1 = NEG_BIG;
      const float tm = rmax16(fmaxf(v0, v1));
      const float nm = fmaxf(rmax[e], tm);
      const float corr = __expf(rmax[e] - nm);
      const float p0 = __expf(v0 - nm);
      const float p1 = __expf(v1 - nm);
      rsum[e] = rsum[e] * corr + rsum16(p0 + p1);
      rmax[e] = nm;
#pragma unroll
      for (int f = 0; f < 4; ++f) cacc[f][e] *= corr;
      const int prow = (e + 8 * hi) * 32;             // C layout -> row-major P
      pw[prow + r]      = (_Float16)p0;
      pw[prow + 16 + r] = (_Float16)p1;
    }

    const v16h ap = frag_ld(pw, 32, 0, 0);            // P as A fragment
#pragma unroll
    for (int f = 0; f < 4; ++f)
      cacc[f] = wmma_f16(ap, frag_ld(Vs, LDT, f * 16, 0), cacc[f]);
  }

  // normalize + write ctx as (B,S,D_MODEL) f16
#pragma unroll
  for (int e = 0; e < 8; ++e) {
    const float inv = 1.f / rsum[e];
    const int qrow = q0 + e + 8 * hi;
#pragma unroll
    for (int f = 0; f < 4; ++f) {
      const int col = h * D_HEAD + f * 16 + r;
      ctx[((size_t)b * SS + qrow) * D_MODEL + col] = (_Float16)(cacc[f][e] * inv);
    }
  }
}

// ---------------------------------------------------------------- kernel 3
// out = ctx(f16) @ Wo^T + residual, written f32.  Register-staged pipeline.
__global__ __launch_bounds__(256) void out_gemm_kernel(
    const _Float16* __restrict__ ctx, const float* __restrict__ Wo,
    const float* __restrict__ resid, float* __restrict__ out) {
  __shared__ _Float16 As[BM * LDT];
  __shared__ _Float16 Bs[BN * LDT];

  const int tid = threadIdx.x;
  const int wave = tid >> 5, lane = tid & 31;
  const int wm = wave >> 1, wn = wave & 1;
  const int r = lane & 15, hi = lane >> 4;
  const int m0 = blockIdx.x * BM, n0 = blockIdx.y * BN;

  v8f acc[2][4] = {};
  v4u rau[2];
  v4h rb[4];

  auto loadAB = [&](int k0) {
#pragma unroll
    for (int t = 0; t < 2; ++t) {        // A: 512 chunks of 8 halves
      const int c = tid + t * 256;
      const int row = c >> 2;
      const int c8 = (c & 3) * 8;
      rau[t] = *(const v4u*)(ctx + (size_t)(m0 + row) * D_MODEL + k0 + c8);
    }
#pragma unroll
    for (int t = 0; t < 4; ++t) {        // B: 1024 float4 chunks
      const int c = tid + t * 256;
      const int row = c >> 3;
      const int c4 = (c & 7) * 4;
      const float4 b4 = *(const float4*)(Wo + (size_t)(n0 + row) * D_MODEL + k0 + c4);
      v4h bh = { (_Float16)b4.x, (_Float16)b4.y, (_Float16)b4.z, (_Float16)b4.w };
      rb[t] = bh;
    }
  };

  loadAB(0);
  for (int k0 = 0; k0 < D_MODEL; k0 += BK) {
    __syncthreads();
#pragma unroll
    for (int t = 0; t < 2; ++t) {
      const int c = tid + t * 256;
      const int row = c >> 2;
      const int c8 = (c & 3) * 8;
      *(v4u*)(As + row * LDT + c8) = rau[t];
    }
#pragma unroll
    for (int t = 0; t < 4; ++t) {
      const int c = tid + t * 256;
      const int row = c >> 3;
      const int c4 = (c & 7) * 4;
      *(v4h*)(Bs + row * LDT + c4) = rb[t];
    }
    __syncthreads();
    if (k0 + BK < D_MODEL) loadAB(k0 + BK);

    v16h a0 = frag_ld(As, LDT, wm * 32, 0);
    v16h a1 = frag_ld(As, LDT, wm * 32 + 16, 0);
#pragma unroll
    for (int j = 0; j < 4; ++j) {
      v16h bf = frag_ld(Bs, LDT, wn * 64 + j * 16, 0);
      acc[0][j] = wmma_f16(a0, bf, acc[0][j]);
      acc[1][j] = wmma_f16(a1, bf, acc[1][j]);
    }
  }

#pragma unroll
  for (int i = 0; i < 2; ++i)
#pragma unroll
    for (int j = 0; j < 4; ++j)
#pragma unroll
      for (int e = 0; e < 8; ++e) {
        const int m = m0 + wm * 32 + i * 16 + e + 8 * hi;
        const int n = n0 + wn * 64 + j * 16 + r;
        out[(size_t)m * D_MODEL + n] =
            acc[i][j][e] + resid[(size_t)m * D_MODEL + n];
      }
}

// ---------------------------------------------------------------- kernel 4
// In-place LayerNorm over the last dim (1024) of out.
__global__ __launch_bounds__(256) void ln_kernel(float* __restrict__ out,
                                                 const float* __restrict__ gamma,
                                                 const float* __restrict__ beta) {
  __shared__ float ssum[256];
  __shared__ float ssq[256];
  const int row = blockIdx.x;
  const int tid = threadIdx.x;
  float x[4];
  float s = 0.f, s2 = 0.f;
#pragma unroll
  for (int i = 0; i < 4; ++i) {
    x[i] = out[(size_t)row * D_MODEL + tid + i * 256];
    s += x[i];
    s2 += x[i] * x[i];
  }
  ssum[tid] = s; ssq[tid] = s2;
  __syncthreads();
  for (int off = 128; off > 0; off >>= 1) {
    if (tid < off) { ssum[tid] += ssum[tid + off]; ssq[tid] += ssq[tid + off]; }
    __syncthreads();
  }
  const float mean = ssum[0] * (1.f / D_MODEL);
  const float var = ssq[0] * (1.f / D_MODEL) - mean * mean;
  const float rstd = rsqrtf(var + LN_EPS);
#pragma unroll
  for (int i = 0; i < 4; ++i) {
    const int c = tid + i * 256;
    out[(size_t)row * D_MODEL + c] = (x[i] - mean) * rstd * gamma[c] + beta[c];
  }
}

// ---------------------------------------------------------------- launch
extern "C" void kernel_launch(void* const* d_in, const int* in_sizes, int n_in,
                              void* d_out, int out_size, void* d_ws, size_t ws_size,
                              hipStream_t stream) {
  (void)in_sizes; (void)n_in; (void)out_size; (void)ws_size;
  const float* q     = (const float*)d_in[0];
  const float* k     = (const float*)d_in[1];
  const float* v     = (const float*)d_in[2];
  const float* Wq    = (const float*)d_in[3];
  const float* Wk    = (const float*)d_in[4];
  const float* Wv    = (const float*)d_in[5];
  const float* Wo    = (const float*)d_in[6];
  const float* gamma = (const float*)d_in[7];
  const float* beta  = (const float*)d_in[8];
  const int*   slen  = (const int*)d_in[9];
  float* out = (float*)d_out;

  // workspace: 4 f16 tensors of B*H*S*d = 4,194,304 halves each (32 MB total)
  const size_t TEN = (size_t)BB * HEADS * SS * D_HEAD;
  _Float16* qh  = (_Float16*)d_ws;
  _Float16* kh  = qh + TEN;
  _Float16* vhT = kh + TEN;
  _Float16* ctx = vhT + TEN;

  dim3 blk(256);
  // 1) QKV projections (z = 0/1/2 -> Q/K/V)
  qkv_gemm_kernel<<<dim3(M_TOT / BM, D_MODEL / BN, 3), blk, 0, stream>>>(
      q, k, v, Wq, Wk, Wv, qh, kh, vhT);
  // 2) flash attention (async K/V staging into LDS)
  attn_kernel<<<dim3(SS / 128, BB * HEADS), blk, 0, stream>>>(
      qh, kh, vhT, slen, ctx);
  // 3) output projection + residual
  out_gemm_kernel<<<dim3(M_TOT / BM, D_MODEL / BN), blk, 0, stream>>>(
      ctx, Wo, q, out);
  // 4) layernorm (in place on d_out)
  ln_kernel<<<dim3(M_TOT), blk, 0, stream>>>(out, gamma, beta);
}
